// RunLengthEventTransformerEmbedding_87050397155898
// MI455X (gfx1250) — matrix operand order repaired and checked by reading, not compiled, not run on hardware
//
#include <hip/hip_runtime.h>
#include <cstdint>
#include <cstddef>

// ---------------------------------------------------------------------------
// Problem constants (match reference setup_inputs)
// ---------------------------------------------------------------------------
#define BATCH 8
#define TLEN  4096
#define CCH   8
#define DMODEL 512
#define FFDIM 2048
#define NLAYER 4
#define NEVENT 1024
#define NHEADS 8
#define DHEAD  64

typedef __bf16 bf16x16 __attribute__((ext_vector_type(16)));
typedef float  f32x8   __attribute__((ext_vector_type(8)));

union Frag16 { bf16x16 v; uint4 q[2]; };

// ---------------------------------------------------------------------------
// Async global->LDS (gfx1250): global_load_async_to_lds_b128 + s_wait_asynccnt.
// Signature (from compiler diagnostic): (v4i AS(1)* src, v4i AS(3)* dst, Ii, Ii)
// ---------------------------------------------------------------------------
#ifndef __has_builtin
#define __has_builtin(x) 0
#endif

typedef int v4i_vs __attribute__((vector_size(16)));
typedef __attribute__((address_space(1))) v4i_vs as1_v4i;
typedef __attribute__((address_space(3))) v4i_vs as3_v4i;

#if __has_builtin(__builtin_amdgcn_global_load_async_to_lds_b128)
#define ASYNC_COPY16(dst, src) \
    __builtin_amdgcn_global_load_async_to_lds_b128((as1_v4i*)(src), (as3_v4i*)(dst), 0, 0)
#if __has_builtin(__builtin_amdgcn_s_wait_asynccnt)
#define ASYNC_WAIT() __builtin_amdgcn_s_wait_asynccnt(0)
#else
#define ASYNC_WAIT() asm volatile("s_wait_asynccnt 0x0" ::: "memory")
#endif
#else
#define ASYNC_COPY16(dst, src) (*(uint4*)(dst) = *(const uint4*)(src))
#define ASYNC_WAIT()
#endif

__device__ inline uint16_t f2bf(float f) {
    union { float f; uint32_t u; } v; v.f = f;
    uint32_t u = v.u;
    uint32_t r = u + 0x7FFFu + ((u >> 16) & 1u);
    return (uint16_t)(r >> 16);
}

__device__ inline float gelu_tanh(float v) {
    const float c = 0.7978845608028654f;
    float t = tanhf(c * (v + 0.044715f * v * v * v));
    return 0.5f * v * (1.0f + t);
}

// ---------------------------------------------------------------------------
// 1) Weight conversion f32 -> bf16, TRANSPOSED: src [L][R][C] -> dst [L][C][R]
// ---------------------------------------------------------------------------
__global__ __launch_bounds__(256) void cvt_bf16_t_kernel(const float* __restrict__ src,
                                                         uint16_t* __restrict__ dst,
                                                         int L, int R, int C) {
    int idx = blockIdx.x * 256 + threadIdx.x;
    int total = L * R * C;
    if (idx >= total) return;
    int l = idx / (R * C);
    int rem = idx - l * (R * C);
    int r = rem / C, c = rem - r * C;
    dst[(size_t)l * R * C + (size_t)c * R + r] = f2bf(src[idx]);
}

// ---------------------------------------------------------------------------
// 2) Run-length event builder. One block per batch, 256 threads.
// ---------------------------------------------------------------------------
__global__ __launch_bounds__(256) void build_events_kernel(const float* __restrict__ x,
                                                           float* __restrict__ rows,
                                                           float* __restrict__ maskb) {
    const int b   = blockIdx.x;
    const int tid = threadIdx.x;
    __shared__ int scan[256];
    const float* xb = x + (size_t)b * TLEN * CCH;
    const int tpt = TLEN / 256;
    const int t0  = tid * tpt;

    int cnt = 0;
    for (int t = t0; t < t0 + tpt; ++t) {
        for (int c = 0; c < CCH; ++c) {
            int cur  = xb[(size_t)t * CCH + c] > 0.5f;
            int prev = (t == 0) ? -1 : (xb[(size_t)(t - 1) * CCH + c] > 0.5f);
            if (cur != prev) ++cnt;
        }
    }
    scan[tid] = cnt;
    __syncthreads();
    for (int off = 1; off < 256; off <<= 1) {
        int v = (tid >= off) ? scan[tid - off] : 0;
        __syncthreads();
        scan[tid] += v;
        __syncthreads();
    }
    int e = scan[tid] - cnt;

    for (int t = t0; t < t0 + tpt; ++t) {
        for (int c = 0; c < CCH; ++c) {
            int cur  = xb[(size_t)t * CCH + c] > 0.5f;
            int prev = (t == 0) ? -1 : (xb[(size_t)(t - 1) * CCH + c] > 0.5f);
            if (cur != prev) {
                if (e < NEVENT) {
                    int tt = t + 1;
                    while (tt < TLEN && ((xb[(size_t)tt * CCH + c] > 0.5f) == (cur == 1))) ++tt;
                    int dur = tt - t;
                    float* r = rows + ((size_t)b * NEVENT + e) * (CCH + 3);
                    r[c]       = 1.0f;
                    r[CCH]     = (float)cur;
                    r[CCH + 1] = (float)t / (float)(TLEN - 1);
                    r[CCH + 2] = (float)dur / (float)TLEN;
                    maskb[(size_t)b * NEVENT + e] = 1.0f;
                }
                ++e;
            }
        }
    }
}

// ---------------------------------------------------------------------------
// 3) Event projection (K=11, plain f32)
// ---------------------------------------------------------------------------
__global__ __launch_bounds__(256) void proj_kernel(const float* __restrict__ rows,
                                                   const float* __restrict__ pw,
                                                   const float* __restrict__ pb,
                                                   float* __restrict__ H) {
    int idx = blockIdx.x * 256 + threadIdx.x;
    const int total = BATCH * NEVENT * DMODEL;
    if (idx >= total) return;
    int r = idx / DMODEL, d = idx % DMODEL;
    const float* rr = rows + (size_t)r * (CCH + 3);
    float acc = pb[d];
#pragma unroll
    for (int k = 0; k < CCH + 3; ++k) acc += rr[k] * pw[k * DMODEL + d];
    H[idx] = acc;
}

// ---------------------------------------------------------------------------
// 4) LayerNorm -> bf16 (one block per row of 512)
// ---------------------------------------------------------------------------
__global__ __launch_bounds__(256) void ln_bf16_kernel(const float* __restrict__ H,
                                                      const float* __restrict__ g,
                                                      const float* __restrict__ bt,
                                                      uint16_t* __restrict__ Z) {
    int row = blockIdx.x, tid = threadIdx.x;
    __shared__ float red[256];
    __shared__ float sm, sv;
    const float* hr = H + (size_t)row * DMODEL;
    float x0 = hr[tid], x1 = hr[tid + 256];
    red[tid] = x0 + x1;
    __syncthreads();
    for (int s = 128; s; s >>= 1) { if (tid < s) red[tid] += red[tid + s]; __syncthreads(); }
    if (tid == 0) sm = red[0] / (float)DMODEL;
    __syncthreads();
    float m = sm, d0 = x0 - m, d1 = x1 - m;
    red[tid] = d0 * d0 + d1 * d1;
    __syncthreads();
    for (int s = 128; s; s >>= 1) { if (tid < s) red[tid] += red[tid + s]; __syncthreads(); }
    if (tid == 0) sv = red[0] / (float)DMODEL;
    __syncthreads();
    float rs = rsqrtf(sv + 1e-5f);
    Z[(size_t)row * DMODEL + tid]       = f2bf(d0 * rs * g[tid] + bt[tid]);
    Z[(size_t)row * DMODEL + tid + 256] = f2bf(d1 * rs * g[tid + 256] + bt[tid + 256]);
}

// ---------------------------------------------------------------------------
// 5) bf16 WMMA GEMM:  C[M,N] = A[M,K] * Bt[N,K]^T (+epilogue)
//    128x128 tile, BK=32, 8 waves x (32x64). Double-buffered async staging.
// ---------------------------------------------------------------------------
#define EP_BIAS 1
#define EP_GELU 2
#define EP_RES  4
#define EP_BF16 8
#define EP_VT   16   // write bf16 output in [b][head][d][e] layout (for V)

template <int EPI>
__global__ __launch_bounds__(256) void gemm_bf16_kernel(const uint16_t* __restrict__ A,
                                                        const uint16_t* __restrict__ Bt,
                                                        const float* __restrict__ bias,
                                                        float* __restrict__ Cf,
                                                        uint16_t* __restrict__ Cb,
                                                        int M, int N, int K) {
    __shared__ __align__(16) uint16_t As[2][128][32];   // [m][k]
    __shared__ __align__(16) uint16_t Bs[2][128][32];   // [n][k]
    const int tid  = threadIdx.x;
    const int wave = tid >> 5;
    const int lane = tid & 31;
    const int lg = lane >> 4, lr = lane & 15;
    const int wm = wave >> 1, wn = wave & 1;
    const int m0 = blockIdx.x * 128, n0 = blockIdx.y * 128;

    const int chunk = tid * 2;
    const int r0 = chunk >> 2, kc0 = (chunk & 3) * 8;
    const int r1 = (chunk + 1) >> 2, kc1 = ((chunk + 1) & 3) * 8;

    f32x8 acc[2][4];
#pragma unroll
    for (int a = 0; a < 2; ++a)
#pragma unroll
        for (int nf = 0; nf < 4; ++nf)
#pragma unroll
            for (int i = 0; i < 8; ++i) acc[a][nf][i] = 0.0f;

    // prologue stage
    ASYNC_COPY16(&As[0][r0][kc0], A  + (size_t)(m0 + r0) * K + kc0);
    ASYNC_COPY16(&As[0][r1][kc1], A  + (size_t)(m0 + r1) * K + kc1);
    ASYNC_COPY16(&Bs[0][r0][kc0], Bt + (size_t)(n0 + r0) * K + kc0);
    ASYNC_COPY16(&Bs[0][r1][kc1], Bt + (size_t)(n0 + r1) * K + kc1);
    ASYNC_WAIT();
    __syncthreads();

    for (int k0 = 0; k0 < K; k0 += 32) {
        const int buf = (k0 >> 5) & 1;
        if (k0 + 32 < K) {            // prefetch next tile into other buffer
            const int kn = k0 + 32;
            ASYNC_COPY16(&As[buf ^ 1][r0][kc0], A  + (size_t)(m0 + r0) * K + kn + kc0);
            ASYNC_COPY16(&As[buf ^ 1][r1][kc1], A  + (size_t)(m0 + r1) * K + kn + kc1);
            ASYNC_COPY16(&Bs[buf ^ 1][r0][kc0], Bt + (size_t)(n0 + r0) * K + kn + kc0);
            ASYNC_COPY16(&Bs[buf ^ 1][r1][kc1], Bt + (size_t)(n0 + r1) * K + kn + kc1);
        }

        Frag16 af[2];
#pragma unroll
        for (int mf = 0; mf < 2; ++mf) {
            const uint4* ar = (const uint4*)&As[buf][wm * 32 + mf * 16 + lr][0];
            af[mf].q[0] = ar[lg];
            af[mf].q[1] = ar[lg + 2];
        }
#pragma unroll
        for (int nf = 0; nf < 4; ++nf) {
            Frag16 bfr;
            const uint4* br = (const uint4*)&Bs[buf][wn * 64 + nf * 16 + lr][0];
            bfr.q[0] = br[lg];
            bfr.q[1] = br[lg + 2];
#pragma unroll
            for (int mf = 0; mf < 2; ++mf)
                acc[mf][nf] = __builtin_amdgcn_wmma_f32_16x16x32_bf16(
                    false, af[mf].v, false, bfr.v, (short)0, acc[mf][nf], false, false);
        }
        ASYNC_WAIT();
        __syncthreads();
    }

#pragma unroll
    for (int mf = 0; mf < 2; ++mf)
#pragma unroll
        for (int nf = 0; nf < 4; ++nf) {
            int col = n0 + wn * 64 + nf * 16 + lr;
            float bv = 0.0f;
            if constexpr (EPI & EP_BIAS) bv = bias[col];
#pragma unroll
            for (int i = 0; i < 8; ++i) {
                int row = m0 + wm * 32 + mf * 16 + lg * 8 + i;
                float v = acc[mf][nf][i] + bv;
                if constexpr (EPI & EP_GELU) v = gelu_tanh(v);
                if constexpr (EPI & EP_BF16) {
                    Cb[(size_t)row * N + col] = f2bf(v);
                } else if constexpr (EPI & EP_VT) {
                    int bi = row >> 10, e = row & (NEVENT - 1);
                    int hh = col >> 6,  d = col & (DHEAD - 1);
                    Cb[(((size_t)bi * NHEADS + hh) * DHEAD + d) * NEVENT + e] = f2bf(v);
                } else if constexpr (EPI & EP_RES) {
                    Cf[(size_t)row * N + col] += v;
                } else {
                    Cf[(size_t)row * N + col] = v;
                }
            }
        }
}

// ---------------------------------------------------------------------------
// 6) Flash attention. Block = (128 queries, head, batch); wave = 16 queries.
// ---------------------------------------------------------------------------
__global__ __launch_bounds__(256) void flash_attn_kernel(const uint16_t* __restrict__ Q,
                                                         const uint16_t* __restrict__ Kb,
                                                         const uint16_t* __restrict__ Vt,
                                                         const float* __restrict__ Mk,
                                                         uint16_t* __restrict__ O) {
    __shared__ __align__(16) uint16_t Ks[2][64][64];     // [key][d]
    __shared__ __align__(16) uint16_t Vs[2][64][64];     // [d][key]
    __shared__ __align__(16) uint16_t Ps[8][16][64];     // per-wave P staging

    const int tid = threadIdx.x;
    const int wave = tid >> 5, lane = tid & 31;
    const int lg = lane >> 4, lr = lane & 15;
    const int qtile = blockIdx.x * 128 + wave * 16;
    const int hidx = blockIdx.y;
    const int hb = hidx * DHEAD;
    const int b  = blockIdx.z;

    const int chunk = tid * 2;
    const int r0 = chunk >> 3, c0 = (chunk & 7) * 8;
    const int r1 = (chunk + 1) >> 3, c1 = ((chunk + 1) & 7) * 8;
    const uint16_t* kbase = Kb + ((size_t)b * NEVENT) * DMODEL + hb;
    const uint16_t* vbase = Vt + ((size_t)b * NHEADS + hidx) * DHEAD * NEVENT;

    Frag16 qa[2];
    {
        const uint16_t* qp = Q + ((size_t)(b * NEVENT + qtile + lr)) * DMODEL + hb;
#pragma unroll
        for (int ck = 0; ck < 2; ++ck) {
            qa[ck].q[0] = *(const uint4*)(qp + ck * 32 + lg * 8);
            qa[ck].q[1] = *(const uint4*)(qp + ck * 32 + 16 + lg * 8);
        }
    }

    f32x8 of[4];
#pragma unroll
    for (int nf = 0; nf < 4; ++nf)
#pragma unroll
        for (int i = 0; i < 8; ++i) of[nf][i] = 0.0f;
    float mrow[8], lrow[8];
#pragma unroll
    for (int i = 0; i < 8; ++i) { mrow[i] = -1e30f; lrow[i] = 0.0f; }

    ASYNC_COPY16(&Ks[0][r0][c0], kbase + (size_t)r0 * DMODEL + c0);
    ASYNC_COPY16(&Ks[0][r1][c1], kbase + (size_t)r1 * DMODEL + c1);
    ASYNC_COPY16(&Vs[0][r0][c0], vbase + (size_t)r0 * NEVENT + c0);
    ASYNC_COPY16(&Vs[0][r1][c1], vbase + (size_t)r1 * NEVENT + c1);
    ASYNC_WAIT();
    __syncthreads();

    for (int kb = 0; kb < NEVENT; kb += 64) {
        const int buf = (kb >> 6) & 1;
        if (kb + 64 < NEVENT) {
            const int kn = kb + 64;
            ASYNC_COPY16(&Ks[buf ^ 1][r0][c0], kbase + (size_t)(kn + r0) * DMODEL + c0);
            ASYNC_COPY16(&Ks[buf ^ 1][r1][c1], kbase + (size_t)(kn + r1) * DMODEL + c1);
            ASYNC_COPY16(&Vs[buf ^ 1][r0][c0], vbase + (size_t)r0 * NEVENT + kn + c0);
            ASYNC_COPY16(&Vs[buf ^ 1][r1][c1], vbase + (size_t)r1 * NEVENT + kn + c1);
        }

        // S = Q K^T / sqrt(dh)
        f32x8 s[4];
#pragma unroll
        for (int nf = 0; nf < 4; ++nf) {
#pragma unroll
            for (int i = 0; i < 8; ++i) s[nf][i] = 0.0f;
#pragma unroll
            for (int ck = 0; ck < 2; ++ck) {
                Frag16 kf;
                const uint4* krw = (const uint4*)&Ks[buf][nf * 16 + lr][0];
                kf.q[0] = krw[ck * 4 + lg];
                kf.q[1] = krw[ck * 4 + lg + 2];
                s[nf] = __builtin_amdgcn_wmma_f32_16x16x32_bf16(
                    false, qa[ck].v, false, kf.v, (short)0, s[nf], false, false);
            }
        }
#pragma unroll
        for (int nf = 0; nf < 4; ++nf) {
            float mv = Mk[(size_t)b * NEVENT + kb + nf * 16 + lr];
            bool valid = mv > 0.0f;
#pragma unroll
            for (int i = 0; i < 8; ++i) {
                float sv = s[nf][i] * 0.125f;
                s[nf][i] = valid ? sv : -1e9f;
            }
        }
        float mnew[8];
#pragma unroll
        for (int i = 0; i < 8; ++i) {
            float mx = mrow[i];
#pragma unroll
            for (int nf = 0; nf < 4; ++nf) mx = fmaxf(mx, s[nf][i]);
#pragma unroll
            for (int off = 1; off <= 8; off <<= 1)
                mx = fmaxf(mx, __shfl_xor(mx, off, 32));
            mnew[i] = mx;
        }
        float alpha[8];
#pragma unroll
        for (int i = 0; i < 8; ++i) {
            alpha[i] = __expf(mrow[i] - mnew[i]);
            lrow[i] *= alpha[i];
            mrow[i]  = mnew[i];
        }
#pragma unroll
        for (int nf = 0; nf < 4; ++nf)
#pragma unroll
            for (int i = 0; i < 8; ++i) of[nf][i] *= alpha[i];

        float psum[8];
#pragma unroll
        for (int i = 0; i < 8; ++i) psum[i] = 0.0f;
#pragma unroll
        for (int nf = 0; nf < 4; ++nf)
#pragma unroll
            for (int i = 0; i < 8; ++i) {
                float p = __expf(s[nf][i] - mnew[i]);
                psum[i] += p;
                Ps[wave][lg * 8 + i][nf * 16 + lr] = f2bf(p);
            }
#pragma unroll
        for (int i = 0; i < 8; ++i) {
#pragma unroll
            for (int off = 1; off <= 8; off <<= 1)
                psum[i] += __shfl_xor(psum[i], off, 32);
            lrow[i] += psum[i];
        }

        // O += P V   (same-wave LDS ops are in-order)
#pragma unroll
        for (int ck = 0; ck < 2; ++ck) {
            Frag16 pa;
            const uint4* pr = (const uint4*)&Ps[wave][lr][0];
            pa.q[0] = pr[ck * 4 + lg];
            pa.q[1] = pr[ck * 4 + lg + 2];
#pragma unroll
            for (int nf = 0; nf < 4; ++nf) {
                Frag16 vf;
                const uint4* vr = (const uint4*)&Vs[buf][nf * 16 + lr][0];
                vf.q[0] = vr[ck * 4 + lg];
                vf.q[1] = vr[ck * 4 + lg + 2];
                of[nf] = __builtin_amdgcn_wmma_f32_16x16x32_bf16(
                    false, pa.v, false, vf.v, (short)0, of[nf], false, false);
            }
        }
        ASYNC_WAIT();
        __syncthreads();
    }

#pragma unroll
    for (int nf = 0; nf < 4; ++nf) {
        int dc = hb + nf * 16 + lr;
#pragma unroll
        for (int i = 0; i < 8; ++i) {
            int row = qtile + lg * 8 + i;
            O[((size_t)(b * NEVENT + row)) * DMODEL + dc] = f2bf(of[nf][i] / lrow[i]);
        }
    }
}

// ---------------------------------------------------------------------------
// 7) Final LN + masked mean pooling. One block per batch.
// ---------------------------------------------------------------------------
__global__ __launch_bounds__(256) void ln_pool_kernel(const float* __restrict__ H,
                                                      const float* __restrict__ msk,
                                                      const float* __restrict__ g,
                                                      const float* __restrict__ bt,
                                                      float* __restrict__ out) {
    const int b = blockIdx.x, tid = threadIdx.x;
    __shared__ float red[256];
    __shared__ float sm, sv;

    float pm = 0.0f;
    for (int e = tid; e < NEVENT; e += 256) pm += msk[(size_t)b * NEVENT + e];
    red[tid] = pm;
    __syncthreads();
    for (int s = 128; s; s >>= 1) { if (tid < s) red[tid] += red[tid + s]; __syncthreads(); }
    float denom = fmaxf(red[0], 1.0f);
    __syncthreads();

    float acc0 = 0.0f, acc1 = 0.0f;
    for (int e = 0; e < NEVENT; ++e) {
        const float* hr = H + ((size_t)b * NEVENT + e) * DMODEL;
        float x0 = hr[tid], x1 = hr[tid + 256];
        red[tid] = x0 + x1;
        __syncthreads();
        for (int s = 128; s; s >>= 1) { if (tid < s) red[tid] += red[tid + s]; __syncthreads(); }
        if (tid == 0) sm = red[0] / (float)DMODEL;
        __syncthreads();
        float m = sm, d0 = x0 - m, d1 = x1 - m;
        red[tid] = d0 * d0 + d1 * d1;
        __syncthreads();
        for (int s = 128; s; s >>= 1) { if (tid < s) red[tid] += red[tid + s]; __syncthreads(); }
        if (tid == 0) sv = red[0] / (float)DMODEL;
        __syncthreads();
        float rs = rsqrtf(sv + 1e-5f);
        float w = msk[(size_t)b * NEVENT + e];
        if (w > 0.0f) {
            acc0 += d0 * rs * g[tid] + bt[tid];
            acc1 += d1 * rs * g[tid + 256] + bt[tid + 256];
        }
        __syncthreads();
    }
    out[(size_t)b * DMODEL + tid]       = acc0 / denom;
    out[(size_t)b * DMODEL + tid + 256] = acc1 / denom;
}

// ---------------------------------------------------------------------------
// Host launcher
// ---------------------------------------------------------------------------
extern "C" void kernel_launch(void* const* d_in, const int* in_sizes, int n_in,
                              void* d_out, int out_size, void* d_ws, size_t ws_size,
                              hipStream_t stream) {
    (void)in_sizes; (void)n_in; (void)out_size; (void)ws_size;

    const float* x      = (const float*)d_in[0];
    const float* proj_w = (const float*)d_in[1];
    const float* proj_b = (const float*)d_in[2];
    const float* wq     = (const float*)d_in[3];
    const float* wk     = (const float*)d_in[4];
    const float* wv     = (const float*)d_in[5];
    const float* wo     = (const float*)d_in[6];
    const float* ln1_g  = (const float*)d_in[7];
    const float* ln1_b  = (const float*)d_in[8];
    const float* w1     = (const float*)d_in[9];
    const float* b1     = (const float*)d_in[10];
    const float* w2     = (const float*)d_in[11];
    const float* b2     = (const float*)d_in[12];
    const float* ln2_g  = (const float*)d_in[13];
    const float* ln2_b  = (const float*)d_in[14];
    const float* lnf_g  = (const float*)d_in[15];
    const float* lnf_b  = (const float*)d_in[16];
    float* out = (float*)d_out;

    const int M = BATCH * NEVENT;          // 8192 rows
    char* base = (char*)d_ws;
    size_t off = 0;
    auto alloc = [&](size_t bytes) -> char* {
        char* p = base + off;
        off += (bytes + 255) & ~(size_t)255;
        return p;
    };

    float*    rows  = (float*)alloc((size_t)M * (CCH + 3) * sizeof(float));
    float*    maskb = (float*)alloc((size_t)M * sizeof(float));
    float*    h     = (float*)alloc((size_t)M * DMODEL * sizeof(float));
    uint16_t* zb    = (uint16_t*)alloc((size_t)M * DMODEL * 2);
    uint16_t* qb    = (uint16_t*)alloc((size_t)M * DMODEL * 2);
    uint16_t* kbuf  = (uint16_t*)alloc((size_t)M * DMODEL * 2);
    uint16_t* vt    = (uint16_t*)alloc((size_t)M * DMODEL * 2);   // [b][h][d][e]
    uint16_t* ab    = (uint16_t*)alloc((size_t)M * DMODEL * 2);
    uint16_t* f1    = (uint16_t*)alloc((size_t)M * FFDIM * 2);
    uint16_t* wq_b  = (uint16_t*)alloc((size_t)NLAYER * DMODEL * DMODEL * 2);
    uint16_t* wk_b  = (uint16_t*)alloc((size_t)NLAYER * DMODEL * DMODEL * 2);
    uint16_t* wv_b  = (uint16_t*)alloc((size_t)NLAYER * DMODEL * DMODEL * 2);
    uint16_t* wo_b  = (uint16_t*)alloc((size_t)NLAYER * DMODEL * DMODEL * 2);
    uint16_t* w1_b  = (uint16_t*)alloc((size_t)NLAYER * DMODEL * FFDIM * 2);
    uint16_t* w2_b  = (uint16_t*)alloc((size_t)NLAYER * FFDIM * DMODEL * 2);

    {
        int n_dd = NLAYER * DMODEL * DMODEL;
        int n_df = NLAYER * DMODEL * FFDIM;
        cvt_bf16_t_kernel<<<(n_dd + 255) / 256, 256, 0, stream>>>(wq, wq_b, NLAYER, DMODEL, DMODEL);
        cvt_bf16_t_kernel<<<(n_dd + 255) / 256, 256, 0, stream>>>(wk, wk_b, NLAYER, DMODEL, DMODEL);
        cvt_bf16_t_kernel<<<(n_dd + 255) / 256, 256, 0, stream>>>(wv, wv_b, NLAYER, DMODEL, DMODEL);
        cvt_bf16_t_kernel<<<(n_dd + 255) / 256, 256, 0, stream>>>(wo, wo_b, NLAYER, DMODEL, DMODEL);
        cvt_bf16_t_kernel<<<(n_df + 255) / 256, 256, 0, stream>>>(w1, w1_b, NLAYER, DMODEL, FFDIM);
        cvt_bf16_t_kernel<<<(n_df + 255) / 256, 256, 0, stream>>>(w2, w2_b, NLAYER, FFDIM, DMODEL);
    }

    hipMemsetAsync(rows, 0, (size_t)M * (CCH + 3) * sizeof(float), stream);
    hipMemsetAsync(maskb, 0, (size_t)M * sizeof(float), stream);
    build_events_kernel<<<BATCH, 256, 0, stream>>>(x, rows, maskb);
    proj_kernel<<<(M * DMODEL + 255) / 256, 256, 0, stream>>>(rows, proj_w, proj_b, h);

    dim3 g_dd(M / 128, DMODEL / 128);    // 64 x 4
    dim3 g_df(M / 128, FFDIM / 128);     // 64 x 16
    dim3 g_at(NEVENT / 128, NHEADS, BATCH);

    for (int l = 0; l < NLAYER; ++l) {
        const size_t odd = (size_t)l * DMODEL * DMODEL;
        const size_t odf = (size_t)l * DMODEL * FFDIM;
        ln_bf16_kernel<<<M, 256, 0, stream>>>(h, ln1_g + l * DMODEL, ln1_b + l * DMODEL, zb);
        gemm_bf16_kernel<EP_BF16><<<g_dd, 256, 0, stream>>>(zb, wq_b + odd, nullptr, nullptr, qb,
                                                            M, DMODEL, DMODEL);
        gemm_bf16_kernel<EP_BF16><<<g_dd, 256, 0, stream>>>(zb, wk_b + odd, nullptr, nullptr, kbuf,
                                                            M, DMODEL, DMODEL);
        gemm_bf16_kernel<EP_VT><<<g_dd, 256, 0, stream>>>(zb, wv_b + odd, nullptr, nullptr, vt,
                                                          M, DMODEL, DMODEL);
        flash_attn_kernel<<<g_at, 256, 0, stream>>>(qb, kbuf, vt, maskb, ab);
        gemm_bf16_kernel<EP_RES><<<g_dd, 256, 0, stream>>>(ab, wo_b + odd, nullptr, h, nullptr,
                                                           M, DMODEL, DMODEL);
        ln_bf16_kernel<<<M, 256, 0, stream>>>(h, ln2_g + l * DMODEL, ln2_b + l * DMODEL, zb);
        gemm_bf16_kernel<EP_BIAS | EP_GELU | EP_BF16><<<g_df, 256, 0, stream>>>(
            zb, w1_b + odf, b1 + l * FFDIM, nullptr, f1, M, FFDIM, DMODEL);
        gemm_bf16_kernel<EP_BIAS | EP_RES><<<g_dd, 256, 0, stream>>>(
            f1, w2_b + odf, b2 + l * DMODEL, h, nullptr, M, DMODEL, FFDIM);
    }

    ln_pool_kernel<<<BATCH, 256, 0, stream>>>(h, maskb, lnf_g, lnf_b, out);
}
